// Model_80436147519514
// MI455X (gfx1250) — matrix operand order repaired and checked by reading, not compiled, not run on hardware
//
#include <hip/hip_runtime.h>

// ---------------------------------------------------------------------------
// DGCNN-style point net for MI455X (gfx1250, wave32, WMMA).
// EdgeConv factored: msg = theta*hj + (phi-theta)*hi + b -> one per-point
// GEMM (bf16 WMMA, f32 accum) + L2-resident gather/max.
// All GEMM operands pre-converted to zero-padded bf16 (Kp = ceil32(K)) so the
// WMMA inner loop is pure b128 loads + v_wmma, no guards, no converts.
// ---------------------------------------------------------------------------

#define BN_S 0.99999500003749968f   // 1/sqrt(1 + 1e-5)

typedef __attribute__((ext_vector_type(16))) __bf16 v16bf;
typedef __attribute__((ext_vector_type(8)))  __bf16 v8bf;
typedef __attribute__((ext_vector_type(8)))  float  v8f;

__device__ __forceinline__ __bf16 f2bf(float f) {
  unsigned u = __builtin_bit_cast(unsigned, f);
  u += 0x7fffu + ((u >> 16) & 1u);             // round-to-nearest-even
  return __builtin_bit_cast(__bf16, (unsigned short)(u >> 16));
}

// fp32 [rows, ld] -> bf16 [rows, Kp], zero-padded K..Kp-1
__global__ __launch_bounds__(256) void cvt_bf16_pad(
    const float* __restrict__ src, int ld, int K, int Kp, int rows,
    __bf16* __restrict__ dst)
{
  const int tid = blockIdx.x * 256 + threadIdx.x;
  if (tid >= rows * Kp) return;
  const int k = tid % Kp;
  const int r = tid / Kp;
  dst[tid] = f2bf((k < K) ? src[(size_t)r * ld + k] : 0.0f);
}

// Fused EdgeConv weights: Wbf = bf16([theta ; phi-theta]) zero-padded to Kp,
// bcat = [0 ; bt+bp] (fp32).
__global__ __launch_bounds__(256) void prep_wcat_bf16(
    const float* __restrict__ Wt, const float* __restrict__ Wp,
    const float* __restrict__ bt, const float* __restrict__ bp,
    int Dout, int Din, int Kp,
    __bf16* __restrict__ Wbf, float* __restrict__ bcat)
{
  const int tid = blockIdx.x * 256 + threadIdx.x;
  const int total = 2 * Dout * Kp;
  if (tid < total) {
    const int r = tid / Kp, k = tid % Kp;
    float v = 0.0f;
    if (k < Din)
      v = (r < Dout) ? Wt[r * Din + k]
                     : (Wp[(r - Dout) * Din + k] - Wt[(r - Dout) * Din + k]);
    Wbf[tid] = f2bf(v);
  }
  if (tid < 2 * Dout)
    bcat[tid] = (tid < Dout) ? 0.0f : (bt[tid - Dout] + bp[tid - Dout]);
}

// ---------------------------------------------------------------------------
// bf16-WMMA GEMM: C[M,N] = A[M,Kp] * W[N,Kp]^T (+bias, +BN, +act)
// A, W packed bf16, Kp % 32 == 0, M and N multiples of 16.
// One wave -> one 16x16 tile, v_wmma_f32_16x16x32_bf16, K-loop unrolled x2.
// act: 0 = none, 1 = leaky_relu(0.2), 2 = relu
// ---------------------------------------------------------------------------
__global__ __launch_bounds__(256) void wmma_gemm_bf16(
    const __bf16* __restrict__ A, const __bf16* __restrict__ W,
    float* __restrict__ C, int ldc,
    int M, int N, int Kp,
    const float* __restrict__ bias,
    const float* __restrict__ bn_g, const float* __restrict__ bn_b,
    int act)
{
  const int wave = threadIdx.x >> 5;
  const int lane = threadIdx.x & 31;
  const int half = lane >> 4;       // 0: lanes 0-15, 1: lanes 16-31
  const int lr   = lane & 15;

  const int mt = blockIdx.x;
  const int nt = blockIdx.y * 8 + wave;
  if (nt * 16 >= N) return;         // uniform per-wave exit, no barriers used

  const int row = mt * 16 + lr;     // A-matrix row held by this lane
  const int col = nt * 16 + lr;     // output column (= W row) held by this lane
  const __bf16* __restrict__ Ap = A + (size_t)row * Kp;
  const __bf16* __restrict__ Bp = W + (size_t)col * Kp;

  v8f acc = {};
#pragma unroll 2
  for (int kk = 0; kk < Kp; kk += 32) {
    // A layout (16x32 bf16): lane half 0 -> K {kk..kk+7, kk+16..kk+23},
    //                        lane half 1 -> +8 on both runs.   (16B loads)
    const v8bf a0 = *(const v8bf*)(Ap + kk + half * 8);
    const v8bf a1 = *(const v8bf*)(Ap + kk + 16 + half * 8);
    // B layout (32x16 bf16): same K order as A (half 0 -> kk..kk+7, kk+16..kk+23; half 1 -> +8)
    const v8bf b0 = *(const v8bf*)(Bp + kk + half * 8);
    const v8bf b1 = *(const v8bf*)(Bp + kk + 16 + half * 8);
    const v16bf a = __builtin_shufflevector(a0, a1, 0,1,2,3,4,5,6,7,8,9,10,11,12,13,14,15);
    const v16bf b = __builtin_shufflevector(b0, b1, 0,1,2,3,4,5,6,7,8,9,10,11,12,13,14,15);
    acc = __builtin_amdgcn_wmma_f32_16x16x32_bf16(
        false, a, false, b, (short)0, acc, false, false);
  }

  const float cbias = bias ? bias[col] : 0.0f;
  const float gs = bn_g ? bn_g[col] * BN_S : 1.0f;
  const float gb = bn_b ? bn_b[col] : 0.0f;
#pragma unroll
  for (int r = 0; r < 8; ++r) {
    const int m = mt * 16 + half * 8 + r;   // C/D layout: VGPR r -> M = 8*half+r
    float v = acc[r] + cbias;
    if (bn_g) v = v * gs + gb;
    if (act == 1) v = (v > 0.0f) ? v : 0.2f * v;
    else if (act == 2) v = fmaxf(v, 0.0f);
    C[(size_t)m * ldc + col] = v;
  }
}

// ---------------------------------------------------------------------------
// Brute-force kNN (K=20, N=1024/batch), D compile-time so query features stay
// in VGPRs. One thread per query; neighbors LDS-tiled 128 points at a time.
// ---------------------------------------------------------------------------
template <int D>
__global__ __launch_bounds__(128) void knn_kernel(
    const float* __restrict__ H, int ldh, int* __restrict__ idx)
{
  constexpr int D4 = (D + 3) & ~3;
  __shared__ float sh[128 * D];                    // <= 64 KB (D=128)
  const int tid = threadIdx.x;
  const int q   = blockIdx.x * 128 + tid;          // global point id
  const int b   = q >> 10;                         // batch (N = 1024)

  float qf[D4];
#pragma unroll
  for (int d = 0; d < D4; ++d)
    qf[d] = (d < D) ? H[(size_t)q * ldh + d] : 0.0f;

  float bd[20]; int bi[20];
#pragma unroll
  for (int k = 0; k < 20; ++k) { bd[k] = 3.4e38f; bi[k] = 0; }

  for (int t = 0; t < 8; ++t) {                    // 8 tiles x 128 points
    for (int j = tid; j < 128 * D; j += 128) {
      const int p = j / D, d = j % D;              // compile-time div
      sh[p * D + d] = H[(size_t)(b * 1024 + t * 128 + p) * ldh + d];
    }
    __syncthreads();
    for (int p = 0; p < 128; ++p) {
      const float* __restrict__ shp = &sh[p * D];
      float s0 = 0.f, s1 = 0.f, s2 = 0.f, s3 = 0.f;   // break FMA chain
#pragma unroll
      for (int d = 0; d < D4; d += 4) {
        const float e0 = qf[d + 0] - ((d + 0 < D) ? shp[d + 0] : 0.0f);
        const float e1 = qf[d + 1] - ((d + 1 < D) ? shp[d + 1] : 0.0f);
        const float e2 = qf[d + 2] - ((d + 2 < D) ? shp[d + 2] : 0.0f);
        const float e3 = qf[d + 3] - ((d + 3 < D) ? shp[d + 3] : 0.0f);
        s0 = fmaf(e0, e0, s0); s1 = fmaf(e1, e1, s1);
        s2 = fmaf(e2, e2, s2); s3 = fmaf(e3, e3, s3);
      }
      const float s = (s0 + s1) + (s2 + s3);
      if (s < bd[19]) {                            // sorted insert, top-20
        int pos = 19;
        while (pos > 0 && bd[pos - 1] > s) {
          bd[pos] = bd[pos - 1]; bi[pos] = bi[pos - 1]; --pos;
        }
        bd[pos] = s; bi[pos] = t * 128 + p;
      }
    }
    __syncthreads();
  }
#pragma unroll
  for (int k = 0; k < 20; ++k) idx[(size_t)q * 20 + k] = bi[k];
}

// h_out[p,c] = leaky( (max_k T[idx[p,k],c] + P[p,c]) * g*bn_s + b )
// TP holds [T | P] with ldtp = 2*Dout; gathers stay in L2 (1 MB slab/batch).
__global__ __launch_bounds__(256) void gathermax_kernel(
    const float* __restrict__ TP, int ldtp, const int* __restrict__ idx,
    const float* __restrict__ g, const float* __restrict__ bb,
    float* __restrict__ Hout, int ldh, int Dout)
{
  const int tid = blockIdx.x * blockDim.x + threadIdx.x;
  if (tid >= 32768 * Dout) return;
  const int c = tid % Dout;
  const int p = tid / Dout;
  const int b = p >> 10;
  const int* __restrict__ ip = idx + (size_t)p * 20;
  const float Pv = TP[(size_t)p * ldtp + Dout + c];
  float m = -3.4e38f;
#pragma unroll 4
  for (int k = 0; k < 20; ++k) {
    const int nb = b * 1024 + ip[k];
    m = fmaxf(m, TP[(size_t)nb * ldtp + c]);
  }
  float v = (m + Pv) * (g[c] * BN_S) + bb[c];
  Hout[(size_t)p * ldh + c] = (v > 0.0f) ? v : 0.2f * v;
}

// pooled[b, 0:1024] = max_n proj ; pooled[b, 1024:2048] = mean_n proj
__global__ __launch_bounds__(256) void pool_kernel(
    const float* __restrict__ X, float* __restrict__ pooled)
{
  const int tid = blockIdx.x * blockDim.x + threadIdx.x;
  if (tid >= 32 * 1024) return;
  const int b = tid >> 10, c = tid & 1023;
  const float* __restrict__ base = X + (size_t)b * 1024 * 1024 + c;
  float mx = -3.4e38f, sm = 0.0f;
  for (int n = 0; n < 1024; ++n) {
    const float v = base[(size_t)n * 1024];
    mx = fmaxf(mx, v); sm += v;
  }
  pooled[b * 2048 + c] = mx;
  pooled[b * 2048 + 1024 + c] = sm * (1.0f / 1024.0f);
}

// Tiny 32->16->8->3 relu head; one thread per batch row; p and n identical.
__global__ void head_kernel(const float* __restrict__ Z,
    const float* __restrict__ l1W, const float* __restrict__ l1b,
    const float* __restrict__ l2W, const float* __restrict__ l2b,
    const float* __restrict__ l3W, const float* __restrict__ l3b,
    float* __restrict__ out)
{
  const int r = threadIdx.x;
  if (r >= 32) return;
  float z[32], h1[16], h2[8];
#pragma unroll
  for (int i = 0; i < 32; ++i) z[i] = Z[r * 32 + i];
#pragma unroll
  for (int o = 0; o < 16; ++o) {
    float s = l1b[o];
#pragma unroll
    for (int i = 0; i < 32; ++i) s = fmaf(l1W[o * 32 + i], z[i], s);
    h1[o] = fmaxf(s, 0.0f);
  }
#pragma unroll
  for (int o = 0; o < 8; ++o) {
    float s = l2b[o];
#pragma unroll
    for (int i = 0; i < 16; ++i) s = fmaf(l2W[o * 16 + i], h1[i], s);
    h2[o] = fmaxf(s, 0.0f);
  }
#pragma unroll
  for (int o = 0; o < 3; ++o) {
    float s = l3b[o];
#pragma unroll
    for (int i = 0; i < 8; ++i) s = fmaf(l3W[o * 8 + i], h2[i], s);
    const float v = fmaxf(s, 0.0f);
    out[r * 3 + o] = v;          // p
    out[96 + r * 3 + o] = v;     // n (shared head)
  }
}

// ---------------------------------------------------------------------------
extern "C" void kernel_launch(void* const* d_in, const int* in_sizes, int n_in,
                              void* d_out, int out_size, void* d_ws, size_t ws_size,
                              hipStream_t stream)
{
  (void)in_sizes; (void)n_in; (void)out_size; (void)ws_size;

  // ---- inputs (setup_inputs dict order, tuples flattened) ----
  const float* x = (const float*)d_in[0];
  const float* cWt[4]; const float* cbt[4]; const float* cWp[4]; const float* cbp[4];
  const float* cg[4];  const float* cb[4];
  for (int i = 0; i < 4; ++i) {
    cWt[i] = (const float*)d_in[1 + i];
    cbt[i] = (const float*)d_in[5 + i];
    cWp[i] = (const float*)d_in[9 + i];
    cbp[i] = (const float*)d_in[13 + i];
    cg[i]  = (const float*)d_in[17 + i];
    cb[i]  = (const float*)d_in[21 + i];
  }
  const float* proj_W = (const float*)d_in[25];
  const float* proj_b = (const float*)d_in[26];
  const float* embW[2] = {(const float*)d_in[27], (const float*)d_in[28]};
  const float* embb[2] = {(const float*)d_in[29], (const float*)d_in[30]};
  const float* bneg[2] = {(const float*)d_in[31], (const float*)d_in[32]};
  const float* bneb[2] = {(const float*)d_in[33], (const float*)d_in[34]};
  const float* po_W = (const float*)d_in[35];
  const float* po_b = (const float*)d_in[36];
  const float* l1W = (const float*)d_in[37]; const float* l1b = (const float*)d_in[38];
  const float* l2W = (const float*)d_in[39]; const float* l2b = (const float*)d_in[40];
  const float* l3W = (const float*)d_in[41]; const float* l3b = (const float*)d_in[42];

  // ---- workspace carve-up (all regions fully written before read) ----
  char* ws = (char*)d_ws;
  size_t off = 0;
  auto carve = [&](size_t bytes) -> void* {
    void* p = ws + off;
    off = (off + bytes + 255) & ~(size_t)255;
    return p;
  };
  float*  hs     = (float*) carve((size_t)32768 * 512 * 4);   // concat feats, ld=512
  float*  big    = (float*) carve((size_t)32768 * 1024 * 4);  // TP / proj out
  int*    idxbuf = (int*)   carve((size_t)32768 * 20 * 4);
  __bf16* Abf    = (__bf16*)carve((size_t)32768 * 512 * 2);   // packed A (<=Kp 512)
  __bf16* Wbf    = (__bf16*)carve((size_t)1024 * 2048 * 2);   // packed W (max emb1)
  float*  bcat   = (float*) carve((size_t)512 * 4);
  float*  pooled = (float*) carve((size_t)32 * 2048 * 4);
  float*  e1     = (float*) carve((size_t)32 * 512 * 4);
  float*  e2     = (float*) carve((size_t)32 * 256 * 4);
  float*  zb     = (float*) carve((size_t)32 * 32 * 4);

  auto cvt = [&](const float* src, int ld, int K, int Kp, int rows, __bf16* dst) {
    const int total = rows * Kp;
    cvt_bf16_pad<<<(total + 255) / 256, 256, 0, stream>>>(src, ld, K, Kp, rows, dst);
  };
  auto gemm = [&](const __bf16* A, const __bf16* W, float* C, int ldc,
                  int M, int N, int Kp, const float* bias,
                  const float* g, const float* bb, int act) {
    dim3 grid(M / 16, (N + 127) / 128);
    wmma_gemm_bf16<<<grid, 256, 0, stream>>>(A, W, C, ldc, M, N, Kp, bias, g, bb, act);
  };

  const int FEAT[4] = {64, 64, 128, 256};
  const int HOFF[4] = {0, 64, 128, 256};

  const float* Hin = x;
  int ldin = 3, Din = 3;

  for (int i = 0; i < 4; ++i) {
    const int Dout = FEAT[i];
    const int Ncat = 2 * Dout;
    const int Kp   = (Din + 31) & ~31;

    // 1) kNN on current features (D known at compile time)
    if (Din == 3)        knn_kernel<3>  <<<256, 128, 0, stream>>>(Hin, ldin, idxbuf);
    else if (Din == 64)  knn_kernel<64> <<<256, 128, 0, stream>>>(Hin, ldin, idxbuf);
    else                 knn_kernel<128><<<256, 128, 0, stream>>>(Hin, ldin, idxbuf);

    // 2) fused bf16 weights [theta ; phi-theta] (padded), bias [0 ; bt+bp]
    {
      const int total = Ncat * Kp;
      prep_wcat_bf16<<<(total + 255) / 256, 256, 0, stream>>>(
          cWt[i], cWp[i], cbt[i], cbp[i], Dout, Din, Kp, Wbf, bcat);
    }

    // 3) A -> packed bf16, then one per-point GEMM -> [T | P]
    cvt(Hin, ldin, Din, Kp, 32768, Abf);
    gemm(Abf, Wbf, big, Ncat, 32768, Ncat, Kp, bcat, nullptr, nullptr, 0);

    // 4) gather T rows by idx, max over K, BN + leaky -> hs[:, HOFF..]
    {
      const int total = 32768 * Dout;
      gathermax_kernel<<<(total + 255) / 256, 256, 0, stream>>>(
          big, Ncat, idxbuf, cg[i], cb[i], hs + HOFF[i], 512, Dout);
    }

    Hin = hs + HOFF[i]; ldin = 512; Din = Dout;
  }

  // projection: [32768,512] x [1024,512]^T -> big [32768,1024]
  cvt(hs, 512, 512, 512, 32768, Abf);
  cvt(proj_W, 512, 512, 512, 1024, Wbf);
  gemm(Abf, Wbf, big, 1024, 32768, 1024, 512, proj_b, nullptr, nullptr, 0);

  // global max + mean pooling -> pooled [32, 2048]
  pool_kernel<<<(32 * 1024 + 255) / 256, 256, 0, stream>>>(big, pooled);

  // embedding MLP (BN + leaky fused in WMMA epilogue)
  cvt(pooled, 2048, 2048, 2048, 32, Abf);
  cvt(embW[0], 2048, 2048, 2048, 512, Wbf);
  gemm(Abf, Wbf, e1, 512, 32, 512, 2048, embb[0], bneg[0], bneb[0], 1);

  cvt(e1, 512, 512, 512, 32, Abf);
  cvt(embW[1], 512, 512, 512, 256, Wbf);
  gemm(Abf, Wbf, e2, 256, 32, 256, 512, embb[1], bneg[1], bneb[1], 1);

  cvt(e2, 256, 256, 256, 32, Abf);
  cvt(po_W, 256, 256, 256, 32, Wbf);
  gemm(Abf, Wbf, zb, 32, 32, 32, 256, po_b, nullptr, nullptr, 0);

  // shared relu head -> (p, n)
  head_kernel<<<1, 32, 0, stream>>>(zb, l1W, l1b, l2W, l2b, l3W, l3b,
                                    (float*)d_out);
}
